// RNN_autoreg_35442070126823
// MI455X (gfx1250) — matrix operand order repaired and checked by reading, not compiled, not run on hardware
//
#include <hip/hip_runtime.h>
#include <hip/hip_fp16.h>

// MI455X (gfx1250) autoregressive level-LSTM.
// 64 workgroups x 16 batch rows; entire recurrence on-WGP.
// Gate GEMMs: v_wmma_f32_16x16x32_f16, weight B-fragments pinned in VGPRs,
// recurrent h in LDS in A-fragment layout, h1(60 levels)+memory in 320KB LDS.
// Activations via hardware v_tanh_f32 (or v_exp_f32+v_rcp_f32 fallback).

typedef __attribute__((ext_vector_type(16))) _Float16 v16h;
typedef __attribute__((ext_vector_type(8)))  float    v8f;

#define B_    1024
#define T_    8
#define L_    60
#define NX_   4
#define NSFC_ 3
#define NY_   4
#define NYSFC_ 3
#define H_    64
#define G_    256
#define NMEM_ 32
#define MB_   16
#define NWAVE 16
#define NTH   (NWAVE * 32)

#if defined(__has_builtin)
#if __has_builtin(__builtin_amdgcn_tanhf)
#define HAS_HW_TANH 1
#endif
#if __has_builtin(__builtin_amdgcn_exp2f)
#define EXP2F(x) __builtin_amdgcn_exp2f(x)
#endif
#if __has_builtin(__builtin_amdgcn_rcpf)
#define RCPF(x) __builtin_amdgcn_rcpf(x)
#endif
#endif
#ifndef EXP2F
#define EXP2F(x) exp2f(x)
#endif
#ifndef RCPF
#define RCPF(x) (1.0f / (x))
#endif

__device__ __forceinline__ float fast_tanh(float x) {
#ifdef HAS_HW_TANH
  return __builtin_amdgcn_tanhf(x);
#else
  // tanh(x) = 1 - 2/(exp2(2x*log2e)+1)
  return 1.0f - 2.0f * RCPF(EXP2F(x * 2.885390081777927f) + 1.0f);
#endif
}
__device__ __forceinline__ float fast_sigm(float x) {
#ifdef HAS_HW_TANH
  return 0.5f * __builtin_amdgcn_tanhf(0.5f * x) + 0.5f;
#else
  return RCPF(1.0f + EXP2F(-1.4426950408889634f * x));
#endif
}

// ---- WMMA 16-bit A-fragment (16x32, MxK) mapping (ISA 7.12.2) ----
__device__ __forceinline__ int aLane(int m, int kk) { return m + (((kk & 15) >> 3) << 4); }
__device__ __forceinline__ int aIdx(int kk)         { return ((kk >> 4) << 3) + (kk & 7); }
__device__ __forceinline__ int aOff(int m, int k) {  // k in [0,64)
  int kg = k >> 5, kk = k & 31;
  return (kg * 32 + aLane(m, kk)) * 16 + aIdx(kk);
}

// ---- WMMA 16-bit B-fragment (32x16, KxN) loader from row-major W[N][Krow] ----
__device__ __forceinline__ v16h loadB(const float* __restrict__ W, int Krow,
                                      int nt, int kg, int lane) {
  int n  = nt * 16 + (lane & 15);
  int k0 = kg * 32 + ((lane >> 4) << 3);
  v16h r;
#pragma unroll
  for (int i = 0; i < 16; ++i) {
    int k = k0 + i + ((i >> 3) << 3);
    float v = (k < Krow) ? W[n * Krow + k] : 0.f;
    r[i] = (_Float16)v;
  }
  return r;
}

__global__ __launch_bounds__(NTH)
void rnn_autoreg_kernel(const float* __restrict__ x_main, const float* __restrict__ x_sfc,
                        const float* __restrict__ W_sfc1, const float* __restrict__ b_sfc1,
                        const float* __restrict__ W_sfc2, const float* __restrict__ b_sfc2,
                        const float* __restrict__ Wih1, const float* __restrict__ Whh1,
                        const float* __restrict__ b1,
                        const float* __restrict__ Wih2, const float* __restrict__ Whh2,
                        const float* __restrict__ b2,
                        const float* __restrict__ W_lat, const float* __restrict__ b_lat,
                        const float* __restrict__ W_out, const float* __restrict__ b_out,
                        const float* __restrict__ W_sfcout, const float* __restrict__ b_sfcout,
                        float* __restrict__ out) {
  extern __shared__ char smem[];
  _Float16* h1A  = (_Float16*)smem;            // 60*1024 halves, A-frag layout / level
  _Float16* memL = h1A + L_ * 1024;            // 60*16*32 halves (latent memory)
  _Float16* hA   = memL + L_ * MB_ * NMEM_;    // 1024 halves: recurrent h, A-frag layout
  _Float16* xinA = hA + 1024;                  // 1024 halves: rnn1 input, A-frag layout
  _Float16* xl   = xinA + 1024;                // 60*16*4 halves: x tile for current t
  float* gates   = (float*)(xl + L_ * MB_ * NX_);  // 16 x 256
  float* cst     = gates + MB_ * G_;               // 16 x 64
  float* lat32   = cst + MB_ * H_;                 // 16 x 32

  const int tid  = threadIdx.x;
  const int lane = tid & 31;
  const int wv   = tid >> 5;
  const int b0   = blockIdx.x * MB_;

  // ---- weight B-fragments -> VGPRs (loaded once, reused 480x) ----
  const v16h wih1k0 = loadB(Wih1, NX_ + NMEM_, wv, 0, lane);
  const v16h wih1k1 = loadB(Wih1, NX_ + NMEM_, wv, 1, lane);
  const v16h whh1k0 = loadB(Whh1, H_, wv, 0, lane);
  const v16h whh1k1 = loadB(Whh1, H_, wv, 1, lane);
  const v16h wih2k0 = loadB(Wih2, H_, wv, 0, lane);
  const v16h wih2k1 = loadB(Wih2, H_, wv, 1, lane);
  const v16h whh2k0 = loadB(Whh2, H_, wv, 0, lane);
  const v16h whh2k1 = loadB(Whh2, H_, wv, 1, lane);
  const v16h wlat0  = loadB(W_lat, H_, wv & 1, 0, lane);
  const v16h wlat1  = loadB(W_lat, H_, wv & 1, 1, lane);

  // ---- loop-invariant per-thread precompute ----
  const int col   = wv * 16 + (lane & 15);   // gate column this lane owns in C-frag
  const int mbase = (lane >> 4) * 8;
  const int gbase = mbase * G_ + col;
  const float bias1 = b1[col];
  const float bias2 = b2[col];
  const float biasl = b_lat[(wv & 1) * 16 + (lane & 15)];

  // elementwise phase: 2 elements per thread, fixed offsets
  int ew_goff[2], ew_coff[2], ew_hoff[2];
#pragma unroll
  for (int s = 0; s < 2; ++s) {
    int e = tid + s * NTH;           // e in [0,1024)
    int m = e >> 6, j = e & 63;
    ew_goff[s] = m * G_ + j;
    ew_coff[s] = m * H_ + j;
    ew_hoff[s] = aOff(m, j);
  }

  // xin-build phase: elements e<576 cover k in [0,36); pad k>=36 is constant 0
  bool xb_act[2], xb_isx[2];
  int  xb_dst[2], xb_src[2];
#pragma unroll
  for (int s = 0; s < 2; ++s) {
    int e = tid + s * NTH;
    xb_act[s] = (e < (NX_ + NMEM_) * MB_);
    int k = e >> 4, m = e & 15;
    xb_dst[s] = aOff(m, k < H_ ? k : 0);
    if (k < NX_) { xb_isx[s] = true;  xb_src[s] = m * NX_ + k; }
    else         { xb_isx[s] = false; xb_src[s] = m * NMEM_ + (k - NX_); }
  }

  // zero latent memory (t=0) and the constant zero-pad region of xinA
  for (int e = tid; e < L_ * MB_ * NMEM_; e += NTH) memL[e] = (_Float16)0.f;
  for (int e = tid; e < 1024; e += NTH) xinA[e] = (_Float16)0.f;
  __syncthreads();

  for (int t = 0; t < T_; ++t) {
    // ---- stage x tile for this timestep (coalesced), surface MLP init ----
    for (int e = tid; e < MB_ * L_ * NX_; e += NTH) {
      int m = e / (L_ * NX_), r = e % (L_ * NX_);      // r = lv*4+k, contiguous in global
      float v = x_main[((size_t)(b0 + m) * T_ + t) * L_ * NX_ + r];
      xl[((r >> 2) * MB_ + m) * NX_ + (r & 3)] = (_Float16)v;
    }
    for (int e = tid; e < MB_ * H_; e += NTH) {
      int m = e >> 6, j = e & 63;
      const float* s = x_sfc + ((size_t)(b0 + m) * T_ + t) * NSFC_;
      float a1 = b_sfc1[j], a2 = b_sfc2[j];
#pragma unroll
      for (int q = 0; q < NSFC_; ++q) {
        a1 += s[q] * W_sfc1[j * NSFC_ + q];
        a2 += s[q] * W_sfc2[j * NSFC_ + q];
      }
      cst[m * H_ + j] = fast_tanh(a2);
      hA[aOff(m, j)] = (_Float16)fast_tanh(a1);
    }
    __syncthreads();
    // pre-build xinA for first rnn1 level (lv=59)
#pragma unroll
    for (int s = 0; s < 2; ++s)
      if (xb_act[s]) {
        _Float16 v = xb_isx[s] ? xl[(L_ - 1) * MB_ * NX_ + xb_src[s]]
                               : memL[(L_ - 1) * MB_ * NMEM_ + xb_src[s]];
        xinA[xb_dst[s]] = v;
      }
    __syncthreads();

    // ---- rnn1: surface -> TOA (levels 59..0), 2 barriers per level ----
    for (int lv = L_ - 1; lv >= 0; --lv) {
      {
        v16h x0 = *(const v16h*)(xinA + lane * 16);
        v16h x1 = *(const v16h*)(xinA + (32 + lane) * 16);
        v16h h0 = *(const v16h*)(hA + lane * 16);
        v16h h1 = *(const v16h*)(hA + (32 + lane) * 16);
        v8f acc = {};
        acc = __builtin_amdgcn_wmma_f32_16x16x32_f16(false, x0, false, wih1k0, (short)0, acc, false, false);
        acc = __builtin_amdgcn_wmma_f32_16x16x32_f16(false, x1, false, wih1k1, (short)0, acc, false, false);
        acc = __builtin_amdgcn_wmma_f32_16x16x32_f16(false, h0, false, whh1k0, (short)0, acc, false, false);
        acc = __builtin_amdgcn_wmma_f32_16x16x32_f16(false, h1, false, whh1k1, (short)0, acc, false, false);
#pragma unroll
        for (int r = 0; r < 8; ++r) gates[gbase + r * G_] = acc[r] + bias1;
      }
      __syncthreads();
      // elementwise LSTM + build next level's xin fragments
#pragma unroll
      for (int s = 0; s < 2; ++s) {
        int go = ew_goff[s];
        float gi = gates[go], gf = gates[go + 64], gg = gates[go + 128], go_ = gates[go + 192];
        float c = fast_sigm(gf) * cst[ew_coff[s]] + fast_sigm(gi) * fast_tanh(gg);
        float h = fast_sigm(go_) * fast_tanh(c);
        cst[ew_coff[s]] = c;
        _Float16 hh = (_Float16)h;
        hA[ew_hoff[s]] = hh;
        h1A[lv * 1024 + ew_hoff[s]] = hh;
      }
      if (lv > 0) {
#pragma unroll
        for (int s = 0; s < 2; ++s)
          if (xb_act[s]) {
            _Float16 v = xb_isx[s] ? xl[(lv - 1) * MB_ * NX_ + xb_src[s]]
                                   : memL[(lv - 1) * MB_ * NMEM_ + xb_src[s]];
            xinA[xb_dst[s]] = v;
          }
      }
      __syncthreads();
    }

    // ---- rnn2 init: zero state ----
    for (int e = tid; e < 1024; e += NTH) { hA[e] = (_Float16)0.f; cst[e] = 0.f; }
    __syncthreads();

    // ---- rnn2: TOA -> surface (levels 0..59) ----
    for (int lv = 0; lv < L_; ++lv) {
      {
        const _Float16* h1l = h1A + lv * 1024;
        v16h x0 = *(const v16h*)(h1l + lane * 16);
        v16h x1 = *(const v16h*)(h1l + (32 + lane) * 16);
        v16h h0 = *(const v16h*)(hA + lane * 16);
        v16h h1 = *(const v16h*)(hA + (32 + lane) * 16);
        v8f acc = {};
        acc = __builtin_amdgcn_wmma_f32_16x16x32_f16(false, x0, false, wih2k0, (short)0, acc, false, false);
        acc = __builtin_amdgcn_wmma_f32_16x16x32_f16(false, x1, false, wih2k1, (short)0, acc, false, false);
        acc = __builtin_amdgcn_wmma_f32_16x16x32_f16(false, h0, false, whh2k0, (short)0, acc, false, false);
        acc = __builtin_amdgcn_wmma_f32_16x16x32_f16(false, h1, false, whh2k1, (short)0, acc, false, false);
#pragma unroll
        for (int r = 0; r < 8; ++r) gates[gbase + r * G_] = acc[r] + bias2;
      }
      __syncthreads();
#pragma unroll
      for (int s = 0; s < 2; ++s) {
        int go = ew_goff[s];
        float gi = gates[go], gf = gates[go + 64], gg = gates[go + 128], go_ = gates[go + 192];
        float c = fast_sigm(gf) * cst[ew_coff[s]] + fast_sigm(gi) * fast_tanh(gg);
        float h = fast_sigm(go_) * fast_tanh(c);
        cst[ew_coff[s]] = c;
        hA[ew_hoff[s]] = (_Float16)h;
      }
      __syncthreads();
      // latent head: lat = h2 @ W_lat^T + b_lat  (waves 0,1)
      if (wv < 2) {
        v16h h0 = *(const v16h*)(hA + lane * 16);
        v16h h1 = *(const v16h*)(hA + (32 + lane) * 16);
        v8f acc = {};
        acc = __builtin_amdgcn_wmma_f32_16x16x32_f16(false, h0, false, wlat0, (short)0, acc, false, false);
        acc = __builtin_amdgcn_wmma_f32_16x16x32_f16(false, h1, false, wlat1, (short)0, acc, false, false);
        int lcol = (wv & 1) * 16 + (lane & 15);
#pragma unroll
        for (int r = 0; r < 8; ++r) {
          float v = acc[r] + biasl;
          int m = mbase + r;
          lat32[m * NMEM_ + lcol] = v;
          memL[(lv * MB_ + m) * NMEM_ + lcol] = (_Float16)v;
        }
      }
      __syncthreads();
      // out = lat @ W_out^T + b_out
      for (int e = tid; e < MB_ * NY_; e += NTH) {
        int m = e >> 2, q = e & 3;
        float a = b_out[q];
#pragma unroll
        for (int p = 0; p < NMEM_; ++p) a += lat32[m * NMEM_ + p] * W_out[q * NMEM_ + p];
        out[(((size_t)(b0 + m) * T_ + t) * L_ + lv) * NY_ + q] = a;
      }
      if (lv == L_ - 1) {
        for (int e = tid; e < MB_ * NYSFC_; e += NTH) {
          int m = e / NYSFC_, q = e % NYSFC_;
          float a = b_sfcout[q];
          for (int k = 0; k < H_; ++k) a += (float)hA[aOff(m, k)] * W_sfcout[q * H_ + k];
          size_t base = (size_t)B_ * T_ * L_ * NY_;
          out[base + ((size_t)(b0 + m) * T_ + t) * NYSFC_ + q] = a;
        }
      }
      __syncthreads();
    }
  }
}

extern "C" void kernel_launch(void* const* d_in, const int* in_sizes, int n_in,
                              void* d_out, int out_size, void* d_ws, size_t ws_size,
                              hipStream_t stream) {
  (void)in_sizes; (void)n_in; (void)out_size; (void)d_ws; (void)ws_size;
  const size_t shmem =
      (size_t)(L_ * 1024 + L_ * MB_ * NMEM_ + 1024 + 1024 + L_ * MB_ * NX_) * sizeof(_Float16) +
      (size_t)(MB_ * G_ + MB_ * H_ + MB_ * NMEM_) * sizeof(float);  // ~219 KB < 320 KB WGP LDS
  (void)hipFuncSetAttribute((const void*)rnn_autoreg_kernel,
                            hipFuncAttributeMaxDynamicSharedMemorySize, (int)shmem);
  rnn_autoreg_kernel<<<B_ / MB_, NTH, shmem, stream>>>(
      (const float*)d_in[0],  (const float*)d_in[1],
      (const float*)d_in[2],  (const float*)d_in[3],
      (const float*)d_in[4],  (const float*)d_in[5],
      (const float*)d_in[6],  (const float*)d_in[7],  (const float*)d_in[8],
      (const float*)d_in[9],  (const float*)d_in[10], (const float*)d_in[11],
      (const float*)d_in[12], (const float*)d_in[13],
      (const float*)d_in[14], (const float*)d_in[15],
      (const float*)d_in[16], (const float*)d_in[17],
      (float*)d_out);
}
